// RoughNeuralSDE_35699768165181
// MI455X (gfx1250) — matrix operand order repaired and checked by reading, not compiled, not run on hardware
//
#include <hip/hip_runtime.h>
#include <hip/hip_bf16.h>
#include <hip/hip_fp16.h>

typedef __attribute__((ext_vector_type(16))) _Float16 v16h;
typedef __attribute__((ext_vector_type(8)))  float    v8f;

#define SDE_EPS 1e-6f

#if __has_builtin(__builtin_amdgcn_tanhf)
  #define FAST_TANH(x) __builtin_amdgcn_tanhf(x)
#elif __has_builtin(__builtin_amdgcn_tanh_f32)
  #define FAST_TANH(x) __builtin_amdgcn_tanh_f32(x)
#else
  #define FAST_TANH(x) tanhf(x)
#endif

// stable softplus(r) + EPS
__device__ __forceinline__ float softplus_eps(float r) {
    float e = __expf(-fabsf(r));
    return fmaxf(r, 0.0f) + __logf(1.0f + e) + SDE_EPS;
}

// One MLP evaluation for a 16-path tile, computed transposed:
//   D(64hidden x 16paths) = W2^T x h1^T  -> D-layout: col = path = lane%16,
//   rows striped over VGPRs -> layer 3 is a pure per-lane dot product.
// Constants live in registers (f32 except the f16 WMMA A operand).
__device__ __forceinline__ float mlp_eval(
    float t, float x,
    const float (&w10)[32], const float (&w1x)[32], const float (&b1)[32],
    const v16h  (&A)[4][2],
    const float (&b2)[32], const float (&w3)[32], float b3)
{
    // ---- layer 1: h1[k] = tanh(t*W1[0][k] + x*W1[1][k] + b1[k]) ----
    v16h Bt0, Bt1;
#pragma unroll
    for (int j = 0; j < 16; ++j) {
        float p0 = fmaf(t, w10[j],      fmaf(x, w1x[j],      b1[j]));
        float p1 = fmaf(t, w10[16 + j], fmaf(x, w1x[16 + j], b1[16 + j]));
        Bt0[j] = (_Float16)FAST_TANH(p0);
        Bt1[j] = (_Float16)FAST_TANH(p1);
    }

    // ---- layer 2 (WMMA, b2 folded in as C) + layer 3 (per-lane dot) ----
    // 4 independent accumulators (one per M-tile) so the tanh->fma chains of
    // different tiles retire independently and overlap the next tile's WMMAs.
    float acc[4];
#pragma unroll
    for (int mt = 0; mt < 4; ++mt) {
        v8f c;
#pragma unroll
        for (int r = 0; r < 8; ++r) c[r] = b2[mt * 8 + r];
        c = __builtin_amdgcn_wmma_f32_16x16x32_f16(false, A[mt][0], false, Bt0,
                                                   (short)0, c, false, false);
        c = __builtin_amdgcn_wmma_f32_16x16x32_f16(false, A[mt][1], false, Bt1,
                                                   (short)0, c, false, false);
        float a = 0.0f;
#pragma unroll
        for (int r = 0; r < 8; ++r)
            a = fmaf(FAST_TANH(c[r]), w3[mt * 8 + r], a);
        acc[mt] = a;
    }
    float s = (acc[0] + acc[1]) + (acc[2] + acc[3]);
    // partner half-wave (lane^16) holds the complementary 32 hidden rows
    s += __shfl_xor(s, 16, 32);
    return s + b3;
}

__global__ void __launch_bounds__(64, 1)
rough_sde_kernel(const float* __restrict__ fbm,  // (B,N)
                 const float* __restrict__ x0,   // (B,1)
                 const float* __restrict__ dW1, const float* __restrict__ db1,
                 const float* __restrict__ dW2, const float* __restrict__ db2,
                 const float* __restrict__ dW3, const float* __restrict__ db3,
                 const float* __restrict__ gW1, const float* __restrict__ gb1,
                 const float* __restrict__ gW2, const float* __restrict__ gb2,
                 const float* __restrict__ gW3, const float* __restrict__ gb3,
                 float* __restrict__ out,        // (B, N+1)
                 int B, int N)
{
    // block = 64 threads = 2 waves sharing one 16-path tile.
    // wave 0 evaluates the drift net, wave 1 the diffusion net.
    const int lane = threadIdx.x & 31;
    const int w    = threadIdx.x >> 5;           // 0 = drift, 1 = diffusion
    const int m    = lane & 15;                  // path within tile
    const int path = blockIdx.x * 16 + m;
    const bool lo  = lane < 16;

    // exchange buffers: [phase][net][path]
    __shared__ float xch[2][2][16];

    const float dt = 1.0f / (float)N;

    // wave-uniform net selection
    const float* W1 = w ? gW1 : dW1;  const float* b1p = w ? gb1 : db1;
    const float* W2 = w ? gW2 : dW2;  const float* b2p = w ? gb2 : db2;
    const float* W3 = w ? gW3 : dW3;  const float* b3p = w ? gb3 : db3;

    // ===== load this wave's net constants into registers =====
    const int kb    = lo ? 0 : 16;   // B-operand K base for this lane
    const int abase = lo ? 0 : 8;    // unused; A operand K base is kb (line above)
    const int rbase = lo ? 0 : 8;    // D-layout row offset

    float w10[32], w1x[32], b1[32];
#pragma unroll
    for (int j = 0; j < 16; ++j) {
        int k0 = kb + j, k1 = 32 + kb + j;
        w10[j]      = W1[k0];       w10[16 + j] = W1[k1];
        w1x[j]      = W1[64 + k0];  w1x[16 + j] = W1[64 + k1];
        b1[j]       = b1p[k0];      b1[16 + j]  = b1p[k1];
    }

    v16h A[4][2];
#pragma unroll
    for (int mt = 0; mt < 4; ++mt)
#pragma unroll
        for (int kt = 0; kt < 2; ++kt) {
            v16h a;
#pragma unroll
            for (int j = 0; j < 16; ++j) {
                int kp  = j + kb;                       // K within 32-wide tile
                int k   = 32 * kt + kp;                 // hidden-in index
                int col = 16 * mt + m;                  // hidden-out index
                a[j] = (_Float16)W2[k * 64 + col];
            }
            A[mt][kt] = a;
        }

    float b2[32], w3[32];
#pragma unroll
    for (int mt = 0; mt < 4; ++mt)
#pragma unroll
        for (int r = 0; r < 8; ++r) {
            int h = 16 * mt + rbase + r;
            b2[mt * 8 + r] = b2p[h];
            w3[mt * 8 + r] = W3[h];
        }
    const float b3 = b3p[0];

    // ===== time stepping =====
    float x = x0[path];
    const size_t orow = (size_t)path * (size_t)(N + 1);
    if (w == 0 && lo) out[orow] = x;

    const float* fbm_p = fbm + (size_t)path * (size_t)N;

#pragma clang loop unroll(disable)
    for (int i = 0; i < N; ++i) {
        float t  = (float)i * dt;
        float tn = t + dt;
        float du = fbm_p[i];

        // ---- predictor: my net at (t, x) ----
        float v0 = mlp_eval(t, x, w10, w1x, b1, A, b2, w3, b3);
        if (lo) xch[0][w][m] = v0;
        __syncthreads();
        float mu0  = xch[0][0][m];
        float sig0 = softplus_eps(xch[0][1][m]);
        float xp   = fmaf(mu0, dt, fmaf(sig0, du, x));

        // ---- corrector: my net at (t+dt, xp) ----
        float v1 = mlp_eval(tn, xp, w10, w1x, b1, A, b2, w3, b3);
        if (lo) xch[1][w][m] = v1;
        __syncthreads();
        float mu1  = xch[1][0][m];
        float sig1 = softplus_eps(xch[1][1][m]);

        // both waves compute the identical update -> x stays replicated
        x = fmaf(0.5f * (mu0 + mu1), dt, fmaf(0.5f * (sig0 + sig1), du, x));
        if (w == 0 && lo) out[orow + i + 1] = x;
    }
}

extern "C" void kernel_launch(void* const* d_in, const int* in_sizes, int n_in,
                              void* d_out, int out_size, void* d_ws, size_t ws_size,
                              hipStream_t stream) {
    const float* fbm = (const float*)d_in[0];
    const float* x0  = (const float*)d_in[1];
    const float* dW1 = (const float*)d_in[2];
    const float* db1 = (const float*)d_in[3];
    const float* dW2 = (const float*)d_in[4];
    const float* db2 = (const float*)d_in[5];
    const float* dW3 = (const float*)d_in[6];
    const float* db3 = (const float*)d_in[7];
    const float* gW1 = (const float*)d_in[8];
    const float* gb1 = (const float*)d_in[9];
    const float* gW2 = (const float*)d_in[10];
    const float* gb2 = (const float*)d_in[11];
    const float* gW3 = (const float*)d_in[12];
    const float* gb3 = (const float*)d_in[13];
    float* out = (float*)d_out;

    int B = in_sizes[1];            // x0 has B elements
    int N = in_sizes[0] / B;        // fbm has B*N elements

    int blocks = (B + 15) / 16;     // one 16-path tile per 2-wave workgroup

    rough_sde_kernel<<<blocks, 64, 0, stream>>>(
        fbm, x0, dW1, db1, dW2, db2, dW3, db3,
        gW1, gb1, gW2, gb2, gW3, gb3, out, B, N);
}
